// EpLSTMCell_20856361189732
// MI455X (gfx1250) — matrix operand order repaired and checked by reading, not compiled, not run on hardware
//
#include <hip/hip_runtime.h>

// EpLSTM scan for MI455X (gfx1250, wave32, WMMA).
//   build_wcat : Wcat[n,k] = bf16( k<DX ? Wx[n,k] : Wh[n,k-DX] )  [5120 x 2048] bf16 (L2-resident, 20MB)
//   cvt_x      : X -> bf16                                         [T*B*DX] (32MB)
//   init_state : bias = bx+bh ; h_bf16 = bf16(h0) ; c = c0
//   per t (sequential, 128 iterations):
//     gemm_step : g[B,5DH] = [x_t | h] * Wcat^T + bias  (bf16 WMMA, fp32 acc)
//                 each wave: 2 M-tiles x 4 N-tiles -> B fragments reused across 2 A fragments
//     pointwise : gates -> c, h (fp32 out + bf16 copy of h for next step's GEMM)

#define TSTEPS 128
#define BATCH  128
#define DX     1024
#define DH     1024
#define NG     (5 * DH)        // 5120
#define KTOT   (DX + DH)       // 2048
#define KCHUNK 256
#define LDSTR  (KCHUNK + 8)    // padded row stride (bf16) -> 528B: rows land on banks 4r..4r+3, conflict-free

typedef __attribute__((ext_vector_type(16))) __bf16 v16bf;
typedef __attribute__((ext_vector_type(8)))  __bf16 v8bf;
typedef __attribute__((ext_vector_type(8)))  float  v8f;

__global__ void build_wcat(const float* __restrict__ Wx, const float* __restrict__ Wh,
                           __bf16* __restrict__ Wcat) {
    size_t i = (size_t)blockIdx.x * blockDim.x + threadIdx.x;   // over NG*KTOT
    size_t n = i / KTOT, k = i % KTOT;
    float v = (k < DX) ? Wx[n * DX + k] : Wh[n * DH + (k - DX)];
    Wcat[i] = (__bf16)v;
}

__global__ void cvt_x(const float* __restrict__ X, __bf16* __restrict__ Xb) {
    size_t i = (size_t)blockIdx.x * blockDim.x + threadIdx.x;   // over T*B*DX
    Xb[i] = (__bf16)X[i];
}

__global__ void init_state(const float* __restrict__ bx, const float* __restrict__ bh,
                           float* __restrict__ bias,
                           const float* __restrict__ h0, const float* __restrict__ c0,
                           __bf16* __restrict__ hbf, float* __restrict__ c) {
    int i = blockIdx.x * blockDim.x + threadIdx.x;              // over B*DH (covers NG too)
    if (i < NG) bias[i] = bx[i] + bh[i];
    if (i < BATCH * DH) { hbf[i] = (__bf16)h0[i]; c[i] = c0[i]; }
}

// One block = 128 threads = 4 waves. Block computes a 32(M) x 256(N) tile of g.
// Each wave owns 2 M-tiles x 4 N-tiles = 8 accumulators; full K=2048 reduction.
__global__ __launch_bounds__(128)
void gemm_step(const __bf16* __restrict__ Xb,    // [T,B,DX] bf16
               const __bf16* __restrict__ hbf,   // [B,DH]   bf16 (prev step's h)
               const __bf16* __restrict__ Wcat,  // [NG,KTOT] bf16, row-major
               const float*  __restrict__ bias,  // [NG]
               float* __restrict__ g,            // [B,NG] fp32
               int t)
{
    __shared__ __bf16 ldsA[32 * LDSTR];          // 32 rows x KCHUNK (+pad) ~ 17KB

    const int tid   = threadIdx.x;
    const int wave  = tid >> 5;
    const int lane  = tid & 31;
    const int half  = lane >> 4;     // 0: lanes 0-15, 1: lanes 16-31
    const int l16   = lane & 15;
    const int mBase = blockIdx.y * 32;
    const int nBase = blockIdx.x * 256 + wave * 64;

    // accumulators init with per-column bias (C/D layout: N = lane%16 across all 8 VGPRs)
    v8f acc[2][4];
#pragma unroll
    for (int j = 0; j < 4; ++j) {
        float bv = bias[nBase + j * 16 + l16];
#pragma unroll
        for (int i = 0; i < 8; ++i) { acc[0][j][i] = bv; acc[1][j][i] = 0.0f; }
    }
    // (bias belongs once per column; second M-tile accumulates from 0 and bias is
    //  folded below at store time via acc[1] += bias) -- simpler: add at store.

    const __bf16* xrow = Xb + (size_t)t * BATCH * DX;

    for (int k0 = 0; k0 < KTOT; k0 += KCHUNK) {
        // ---- cooperative global -> LDS stage of A chunk (32 rows x KCHUNK bf16) ----
        const __bf16* src;
        size_t rstride;
        if (k0 < DX) { src = xrow + k0;        rstride = DX; }   // x_t half of K
        else         { src = hbf + (k0 - DX);  rstride = DH; }   // h half of K
#pragma unroll
        for (int it = 0; it < 8; ++it) {                 // 32*256/8 = 1024 groups / 128 threads
            int e  = it * 128 + tid;
            int r  = e >> 5;                             // 32 groups of 8 bf16 per row
            int kk = (e & 31) * 8;
            *(v8bf*)(&ldsA[r * LDSTR + kk]) =
                *(const v8bf*)(src + (size_t)(mBase + r) * rstride + kk);
        }
        __syncthreads();

        // prefetch next K-chunk of this wave's B rows (keeps WGP$ warm; Wcat is L2-resident)
        if (k0 + KCHUNK < KTOT)
            __builtin_prefetch(Wcat + (size_t)(nBase + l16) * KTOT + (k0 + KCHUNK), 0, 0);

        // ---- 8 WMMA K-steps per chunk; 2 A frags x 4 B frags = 8 wmma per step ----
#pragma unroll
        for (int kk = 0; kk < KCHUNK; kk += 32) {
            // A fragments (contiguous K order matching B: lanes 0-15 hold K 0..15, lanes 16-31 K 16..31)
            union { v16bf v; v8bf h[2]; } afrag[2];
#pragma unroll
            for (int m = 0; m < 2; ++m) {
                const __bf16* arow = &ldsA[(m * 16 + l16) * LDSTR + kk];
                afrag[m].h[0] = *(const v8bf*)(arow + half * 16);
                afrag[m].h[1] = *(const v8bf*)(arow + half * 16 + 8);
            }
#pragma unroll
            for (int j = 0; j < 4; ++j) {
                // B fragment: lane = column (row of Wcat), halves hold 16 consecutive K
                int col = nBase + j * 16 + l16;
                v16bf bfrag = *(const v16bf*)(Wcat + (size_t)col * KTOT + (k0 + kk) + half * 16);
                acc[0][j] = __builtin_amdgcn_wmma_f32_16x16x32_bf16(
                    false, afrag[0].v, false, bfrag, (short)0, acc[0][j], false, false);
                acc[1][j] = __builtin_amdgcn_wmma_f32_16x16x32_bf16(
                    false, afrag[1].v, false, bfrag, (short)0, acc[1][j], false, false);
            }
        }
        __syncthreads();
    }

    // store g (C/D layout: VGPR i -> M = i + half*8, N = l16); fold bias into second M-tile
#pragma unroll
    for (int j = 0; j < 4; ++j) {
        int col = nBase + j * 16 + l16;
        float bv = bias[col];
#pragma unroll
        for (int i = 0; i < 8; ++i) {
            int m = half * 8 + i;
            g[(size_t)(mBase + m) * NG + col]      = acc[0][j][i];
            g[(size_t)(mBase + 16 + m) * NG + col] = acc[1][j][i] + bv;
        }
    }
}

__device__ __forceinline__ float sigmoidf(float x) { return 1.0f / (1.0f + __expf(-x)); }

__global__ void pointwise_step(const float* __restrict__ g,         // [B,NG]
                               const float* __restrict__ memories,  // [T,B,DH]
                               float* __restrict__ c,               // [B,DH]
                               __bf16* __restrict__ hbf,            // [B,DH]
                               float* __restrict__ out,             // [T,B,DH]
                               int t)
{
    int idx = blockIdx.x * blockDim.x + threadIdx.x;                // over B*DH
    int b = idx / DH, d = idx % DH;
    const float* gr = g + (size_t)b * NG;
    float Xf = gr[DH + d];
    float Xg = gr[2 * DH + d];
    float Xo = gr[3 * DH + d];
    float Xr = gr[4 * DH + d];
    float ft = sigmoidf(Xf);
    float ot = sigmoidf(Xo);
    float it = 1.0f - ft;                                           // tied forget gate
    float gt = tanhf(Xg);
    float rt = sigmoidf(Xr);
    float mt = memories[((size_t)t * BATCH + b) * DH + d];
    float ct = ft * c[idx] + it * gt + rt * tanhf(mt);
    float ht = ot * tanhf(ct);
    c[idx] = ct;
    out[((size_t)t * BATCH + b) * DH + d] = ht;
    hbf[idx] = (__bf16)ht;                                          // feed next step's WMMA
}

extern "C" void kernel_launch(void* const* d_in, const int* in_sizes, int n_in,
                              void* d_out, int out_size, void* d_ws, size_t ws_size,
                              hipStream_t stream) {
    const float* inputs   = (const float*)d_in[0];
    const float* memories = (const float*)d_in[1];
    const float* h0       = (const float*)d_in[2];
    const float* c0       = (const float*)d_in[3];
    const float* Wx       = (const float*)d_in[4];
    const float* bx       = (const float*)d_in[5];
    const float* Wh       = (const float*)d_in[6];
    const float* bh       = (const float*)d_in[7];
    float* out = (float*)d_out;

    // workspace carve-up (all 16B aligned): ~58 MB total
    char* ws = (char*)d_ws;
    __bf16* Wcat = (__bf16*)ws;  ws += (size_t)NG * KTOT * 2;            // 20 MB
    float*  bias = (float*)ws;   ws += (size_t)NG * 4;                   // 20 KB
    __bf16* Xb   = (__bf16*)ws;  ws += (size_t)TSTEPS * BATCH * DX * 2;  // 32 MB
    __bf16* hbf  = (__bf16*)ws;  ws += (size_t)BATCH * DH * 2;           // 256 KB
    float*  c    = (float*)ws;   ws += (size_t)BATCH * DH * 4;           // 512 KB
    float*  g    = (float*)ws;   ws += (size_t)BATCH * NG * 4;           // 2.5 MB

    build_wcat<<<(NG * KTOT) / 256, 256, 0, stream>>>(Wx, Wh, Wcat);
    cvt_x<<<(TSTEPS * BATCH * DX) / 256, 256, 0, stream>>>(inputs, Xb);
    init_state<<<(BATCH * DH) / 256, 256, 0, stream>>>(bx, bh, bias, h0, c0, hbf, c);

    dim3 ggrid(NG / 256, BATCH / 32);   // 20 x 4 blocks, 4 waves each
    for (int t = 0; t < TSTEPS; ++t) {
        gemm_step<<<ggrid, 128, 0, stream>>>(Xb, hbf, Wcat, bias, g, t);
        pointwise_step<<<(BATCH * DH) / 256, 256, 0, stream>>>(g, memories, c, hbf, out, t);
    }
}
